// SelfAttention_86199993631018
// MI455X (gfx1250) — compile-verified, matrix operand order checked by reading, not run on hardware
//
#include <hip/hip_runtime.h>

// Self-attention, B=4, N=4096, D_IN=512, D_H=64, fp32 in/out.
// f16 WMMA (v_wmma_f32_16x16x32_f16) with fp32 accumulation, flash-attention
// streaming softmax.
//
// Round-3 change: attention computes S^T = K Q^T instead of Q K^T.  In the
// C/D layout each lane then owns ONE query column: softmax stats are
// lane-scalar (1 shuffle per reduction instead of 32), P^T in registers is
// directly the B-fragment of out^T = V^T P^T (no LDS transpose round-trip),
// and output stores are contiguous 32B per lane.

#define BB  4
#define NN  4096
#define DIN 512
#define DH  64

typedef __attribute__((ext_vector_type(16))) _Float16 v16h;
typedef __attribute__((ext_vector_type(8)))  _Float16 v8h;
typedef __attribute__((ext_vector_type(8)))  float    v8f;
typedef __attribute__((ext_vector_type(4)))  float    v4f;

static __device__ __forceinline__ v8f wmma16(v16h a, v16h b, v8f c) {
  // (neg_a, A, neg_b, B, c_mod, C, reuse_a, reuse_b)
  return __builtin_amdgcn_wmma_f32_16x16x32_f16(false, a, false, b, (short)0, c,
                                                false, false);
}

static __device__ __forceinline__ v16h mk16(v8h lo, v8h hi) {
  v16h r;
#pragma unroll
  for (int i = 0; i < 8; ++i) { r[i] = lo[i]; r[i + 8] = hi[i]; }
  return r;
}

// Raw fragment halves.
struct rawh { v8h lo, hi; };

// A/B fragment source pattern (ISA 7.12.2): row = lane%16, h = lane/16;
// halves[0..7]  = row[koff +      h*8 + 0..7]
// halves[8..15] = row[koff + 16 + h*8 + 0..7]
static __device__ __forceinline__ rawh load_raw_h(const _Float16* row, int koff, int h) {
  rawh r;
  r.lo = *(const v8h*)(row + koff + h * 8);
  r.hi = *(const v8h*)(row + koff + 16 + h * 8);
  return r;
}

// ---------------------------------------------------------------------------
// Kernel 0: element-wise fp32 -> f16 (8 elements / thread, b128 in+out).
// ---------------------------------------------------------------------------
__global__ __launch_bounds__(256) void cvt16_kernel(
    const float* __restrict__ in, _Float16* __restrict__ out, int n8)
{
  const int i = blockIdx.x * blockDim.x + threadIdx.x;
  if (i >= n8) return;
  const v4f a = *(const v4f*)(in + (size_t)i * 8);
  const v4f b = *(const v4f*)(in + (size_t)i * 8 + 4);
  v8h o;
#pragma unroll
  for (int j = 0; j < 4; ++j) { o[j] = (_Float16)a[j]; o[j + 4] = (_Float16)b[j]; }
  *(v8h*)(out + (size_t)i * 8) = o;
}

// ---------------------------------------------------------------------------
// Kernel 1: Q/K/V projection, one matrix per wave, pure f16 WMMA with
// double-buffered fragments.  Q scaled by 1/sqrt(DH).  V stored transposed:
// vt[b][d][n].
// ---------------------------------------------------------------------------
__global__ __launch_bounds__(128) void qkv_proj_kernel(
    const _Float16* __restrict__ xh,
    const _Float16* __restrict__ wh,     // [3][DH][DIN] f16 (Q,K,V packed)
    _Float16* __restrict__ qo, _Float16* __restrict__ ko,
    _Float16* __restrict__ vt)
{
  const int lane = threadIdx.x & 31;
  const int lr   = lane & 15;
  const int h    = lane >> 4;
  const int wave = blockIdx.x * 4 + (threadIdx.x >> 5);   // 0..3071
  const int mat  = wave >> 10;                            // 0:Q 1:K 2:V
  const int t    = wave & 1023;
  const int b    = t >> 8;                                // 256 tiles per batch
  const int row0 = (t & 255) << 4;

  v8f acc[4];
#pragma unroll
  for (int f = 0; f < 4; ++f) { v8f z = {}; acc[f] = z; }

  const _Float16* xrow = xh + (size_t)(b * NN + row0 + lr) * DIN;
  const _Float16* wbase = wh + (size_t)mat * DH * DIN;
  const _Float16* wr0 = wbase + (size_t)(0 * 16 + lr) * DIN;
  const _Float16* wr1 = wbase + (size_t)(1 * 16 + lr) * DIN;
  const _Float16* wr2 = wbase + (size_t)(2 * 16 + lr) * DIN;
  const _Float16* wr3 = wbase + (size_t)(3 * 16 + lr) * DIN;

  // Preload k0 = 0 fragments.
  rawh a  = load_raw_h(xrow, 0, h);
  rawh b0 = load_raw_h(wr0, 0, h);
  rawh b1 = load_raw_h(wr1, 0, h);
  rawh b2 = load_raw_h(wr2, 0, h);
  rawh b3 = load_raw_h(wr3, 0, h);

#pragma unroll 1
  for (int k0 = 0; k0 < DIN; k0 += 32) {
    // Prefetch next K-step fragments (loads in flight during the WMMAs).
    const int kn = (k0 + 32 < DIN) ? k0 + 32 : 0;
    rawh an  = load_raw_h(xrow, kn, h);
    rawh bn0 = load_raw_h(wr0, kn, h);
    rawh bn1 = load_raw_h(wr1, kn, h);
    rawh bn2 = load_raw_h(wr2, kn, h);
    rawh bn3 = load_raw_h(wr3, kn, h);

    const v16h af = mk16(a.lo, a.hi);
    acc[0] = wmma16(af, mk16(b0.lo, b0.hi), acc[0]);
    acc[1] = wmma16(af, mk16(b1.lo, b1.hi), acc[1]);
    acc[2] = wmma16(af, mk16(b2.lo, b2.hi), acc[2]);
    acc[3] = wmma16(af, mk16(b3.lo, b3.hi), acc[3]);

    a = an; b0 = bn0; b1 = bn1; b2 = bn2; b3 = bn3;
  }

  // C/D layout: VGPR r -> row r + 8*h, column = lane%16 within the 16-tile.
  if (mat == 0) {
#pragma unroll
    for (int f = 0; f < 4; ++f)
#pragma unroll
      for (int r = 0; r < 8; ++r)
        qo[(size_t)(b * NN + row0 + r + 8 * h) * DH + f * 16 + lr] =
            (_Float16)(acc[f][r] * 0.125f);
  } else if (mat == 1) {
#pragma unroll
    for (int f = 0; f < 4; ++f)
#pragma unroll
      for (int r = 0; r < 8; ++r)
        ko[(size_t)(b * NN + row0 + r + 8 * h) * DH + f * 16 + lr] =
            (_Float16)acc[f][r];
  } else {
    // V transposed: fixed d, 8 consecutive rows -> one b128 store.
#pragma unroll
    for (int f = 0; f < 4; ++f) {
      v8h pv;
#pragma unroll
      for (int r = 0; r < 8; ++r) pv[r] = (_Float16)acc[f][r];
      *(v8h*)(vt + (size_t)(b * DH + f * 16 + lr) * NN + row0 + 8 * h) = pv;
    }
  }
}

// ---------------------------------------------------------------------------
// Kernel 2: streaming (flash) attention, transposed-score formulation.
// One wave owns 16 query rows; iterates over key blocks of 32.
//   S^T  = K_tile (A) x Q^T (B)    -> lane = one query column
//   out^T = V^T (A)  x P^T (B)     -> P^T is already B-fragment-shaped
// ---------------------------------------------------------------------------
__global__ __launch_bounds__(128) void flash_attn_kernel(
    const _Float16* __restrict__ qo, const _Float16* __restrict__ ko,
    const _Float16* __restrict__ vt, float* __restrict__ out)
{
  const int lane = threadIdx.x & 31;
  const int lr   = lane & 15;
  const int h    = lane >> 4;
  const int wave = blockIdx.x * 4 + (threadIdx.x >> 5);
  const int b    = wave >> 8;
  const int q0   = (wave & 255) << 4;

  // Q fragments (used as B: B^T row n = Q row q0+n, contiguous).
  const _Float16* qrow = qo + (size_t)(b * NN + q0 + lr) * DH;
  const rawh qr0 = load_raw_h(qrow, 0, h);
  const rawh qr1 = load_raw_h(qrow, 32, h);
  const v16h qb0 = mk16(qr0.lo, qr0.hi);   // d = 0..31
  const v16h qb1 = mk16(qr1.lo, qr1.hi);   // d = 32..63

  v8f acc[4];                              // out^T tiles: row d=f*16+r+8h, col q=lr
#pragma unroll
  for (int f = 0; f < 4; ++f) { v8f z = {}; acc[f] = z; }
  float mrow = -3.0e38f;                   // per-lane (per-query) running max
  float lsum = 0.0f;                       // per-lane (per-query) running sum

  const _Float16* kbase = ko + (size_t)(b * NN) * DH;
  const _Float16* vbase = vt + (size_t)(b * DH) * NN;

  // Preload K fragments for j0 = 0 (A-matrix: row = key, contiguous).
  rawh k00, k01, k10, k11;
  {
    const _Float16* kr0 = kbase + (size_t)lr * DH;
    const _Float16* kr1 = kr0 + 16 * DH;
    k00 = load_raw_h(kr0, 0, h);  k01 = load_raw_h(kr0, 32, h);
    k10 = load_raw_h(kr1, 0, h);  k11 = load_raw_h(kr1, 32, h);
  }

#pragma unroll 1
  for (int j0 = 0; j0 < NN; j0 += 32) {
    // --- S^T = K Q^T for 32 keys (two 16-key subtiles, K-dim 64 chained) ---
    v8f st0 = {}, st1 = {};
    st0 = wmma16(mk16(k00.lo, k00.hi), qb0, st0);
    st0 = wmma16(mk16(k01.lo, k01.hi), qb1, st0);
    st1 = wmma16(mk16(k10.lo, k10.hi), qb0, st1);
    st1 = wmma16(mk16(k11.lo, k11.hi), qb1, st1);
    // lane (q=lr, h): st0[r] = S[q][j0+8h+r], st1[r] = S[q][j0+16+8h+r]

    // --- prefetch V^T fragments for THIS block (consumed after softmax) ---
    rawh vr0 = load_raw_h(vbase + (size_t)(0 * 16 + lr) * NN + j0, 0, h);
    rawh vr1 = load_raw_h(vbase + (size_t)(1 * 16 + lr) * NN + j0, 0, h);
    rawh vr2 = load_raw_h(vbase + (size_t)(2 * 16 + lr) * NN + j0, 0, h);
    rawh vr3 = load_raw_h(vbase + (size_t)(3 * 16 + lr) * NN + j0, 0, h);

    // --- prefetch K fragments for NEXT block ---
    const int jn = (j0 + 32 < NN) ? (j0 + 32) : 0;
    const _Float16* kn0 = kbase + (size_t)(jn + lr) * DH;
    const _Float16* kn1 = kn0 + 16 * DH;
    rawh nk00 = load_raw_h(kn0, 0, h), nk01 = load_raw_h(kn0, 32, h);
    rawh nk10 = load_raw_h(kn1, 0, h), nk11 = load_raw_h(kn1, 32, h);

    // --- online softmax: per-lane query stats, one xor-16 shuffle each ---
    float mx = fmaxf(st0[0], st1[0]);
#pragma unroll
    for (int r = 1; r < 8; ++r) mx = fmaxf(mx, fmaxf(st0[r], st1[r]));
    mx = fmaxf(mx, __shfl_xor(mx, 16, 32));       // other half's 16 keys

    const float nm   = fmaxf(mrow, mx);
    const float corr = __expf(mrow - nm);
    mrow = nm;

    v16h pb;                                      // P^T B-fragment (f16)
    float sum = 0.0f;
#pragma unroll
    for (int r = 0; r < 8; ++r) {
      const float e0 = __expf(st0[r] - nm);
      const float e1 = __expf(st1[r] - nm);
      sum += e0 + e1;
      pb[r]     = (_Float16)e0;                   // keys  8h + r
      pb[r + 8] = (_Float16)e1;                   // keys 16 + 8h + r
    }
    sum += __shfl_xor(sum, 16, 32);
    lsum = lsum * corr + sum;

#pragma unroll
    for (int f = 0; f < 4; ++f)
#pragma unroll
      for (int r = 0; r < 8; ++r) acc[f][r] *= corr;

    // --- out^T += V^T P^T ---
    acc[0] = wmma16(mk16(vr0.lo, vr0.hi), pb, acc[0]);
    acc[1] = wmma16(mk16(vr1.lo, vr1.hi), pb, acc[1]);
    acc[2] = wmma16(mk16(vr2.lo, vr2.hi), pb, acc[2]);
    acc[3] = wmma16(mk16(vr3.lo, vr3.hi), pb, acc[3]);

    // rotate prefetched K fragments
    k00 = nk00; k01 = nk01; k10 = nk10; k11 = nk11;
  }

  // --- normalize and store: lane (q=lr, h) holds d = f*16+8h+0..7 ---
  const float inv = 1.0f / lsum;
  float* obase = out + (size_t)(b * NN + q0 + lr) * DH + 8 * h;
#pragma unroll
  for (int f = 0; f < 4; ++f) {
    v4f o0, o1;
#pragma unroll
    for (int i = 0; i < 4; ++i) { o0[i] = acc[f][i] * inv; o1[i] = acc[f][i + 4] * inv; }
    *(v4f*)(obase + f * 16)     = o0;
    *(v4f*)(obase + f * 16 + 4) = o1;
  }
}

// ---------------------------------------------------------------------------
extern "C" void kernel_launch(void* const* d_in, const int* in_sizes, int n_in,
                              void* d_out, int out_size, void* d_ws, size_t ws_size,
                              hipStream_t stream) {
  const float* x  = (const float*)d_in[0];
  const float* Wq = (const float*)d_in[1];
  const float* Wk = (const float*)d_in[2];
  const float* Wv = (const float*)d_in[3];

  // Workspace layout (f16):
  _Float16* q  = (_Float16*)d_ws;                   // [B][N][DH]   (pre-scaled)
  _Float16* k  = q  + (size_t)BB * NN * DH;         // [B][N][DH]
  _Float16* vt = k  + (size_t)BB * NN * DH;         // [B][DH][N]   (transposed)
  _Float16* xh = vt + (size_t)BB * NN * DH;         // [B][N][DIN]
  _Float16* wh = xh + (size_t)BB * NN * DIN;        // [3][DH][DIN] (Q,K,V)
  float* out = (float*)d_out;

  // 0) fp32 -> f16 pre-pass (x and the three weight matrices).
  const int n8x = BB * NN * DIN / 8;                // 1,048,576
  const int n8w = DH * DIN / 8;                     // 4,096
  cvt16_kernel<<<dim3((n8x + 255) / 256), dim3(256), 0, stream>>>(x, xh, n8x);
  cvt16_kernel<<<dim3((n8w + 255) / 256), dim3(256), 0, stream>>>(Wq, wh + 0 * DH * DIN, n8w);
  cvt16_kernel<<<dim3((n8w + 255) / 256), dim3(256), 0, stream>>>(Wk, wh + 1 * DH * DIN, n8w);
  cvt16_kernel<<<dim3((n8w + 255) / 256), dim3(256), 0, stream>>>(Wv, wh + 2 * DH * DIN, n8w);

  // 1) fused QKV projection (f16 WMMA).
  qkv_proj_kernel<<<dim3(768), dim3(128), 0, stream>>>(xh, wh, q, k, vt);

  // 2) flash attention (f16 WMMA, fp32 online softmax, transposed scores).
  flash_attn_kernel<<<dim3(256), dim3(128), 0, stream>>>(q, k, vt, out);
}
